// Model_7344394076574
// MI455X (gfx1250) — matrix operand order repaired and checked by reading, not compiled, not run on hardware
//
#include <hip/hip_runtime.h>
#include <hip/hip_bf16.h>

typedef unsigned short u16;
typedef unsigned int u32x4 __attribute__((ext_vector_type(4)));
typedef __bf16 v16bf __attribute__((ext_vector_type(16)));
typedef float v8f __attribute__((ext_vector_type(8)));

union FragBF { u32x4 q[2]; v16bf f; };

#define BV 16
#define EDIM 768
#define NHEAD 12
#define HDIM 64
#define SDIM 41
#define LDIM 1681            // 41*41
#define MROWS 26896          // 16*1681
#define MLPDIM 3072
#define QKVDIM 2304
#define CONVK 1216           // 1200 padded to mult of 32

__device__ __forceinline__ u16 f2bf(float f) {
  unsigned u = __float_as_uint(f);
  unsigned r = u + 0x7FFFu + ((u >> 16) & 1u);
  return (u16)(r >> 16);
}

// Async DMA: global -> LDS, 16B per lane, tracked by ASYNCcnt (CDNA5 TDM-lite
// path). VDST supplies the LDS byte offset, GV addressing (saddr = off).
__device__ __forceinline__ void async_ld_b128(const u16* gsrc, u16* ldst) {
  unsigned loff = (unsigned)(unsigned long long)(void*)ldst;  // LDS byte offset
  asm volatile("global_load_async_to_lds_b128 %0, %1, off"
               :: "v"(loff), "v"(gsrc)
               : "memory");
}
__device__ __forceinline__ void wait_async0() {
  asm volatile("s_wait_asynccnt 0" ::: "memory");
}

// ---------------------------------------------------------------------------
// Weight convert: W (layers, K, N) f32  ->  Wt (layers, N, K) bf16
// ---------------------------------------------------------------------------
__global__ void wt_convert(const float* __restrict__ W, u16* __restrict__ Wt,
                           int Kd, int Nd, long total) {
  long idx = (long)blockIdx.x * blockDim.x + threadIdx.x;
  if (idx >= total) return;
  long NK = (long)Nd * Kd;
  long l = idx / NK;
  long rem = idx - l * NK;
  int n = (int)(rem / Kd);
  int k = (int)(rem - (long)n * Kd);
  Wt[idx] = f2bf(W[l * NK + (size_t)k * Nd + n]);
}

// conv_w (768, 12*10*10=1200) is already N-major; pad K 1200->1216 with zeros
__global__ void convwt_convert(const float* __restrict__ W, u16* __restrict__ Wt) {
  int idx = blockIdx.x * 256 + threadIdx.x;
  if (idx >= EDIM * CONVK) return;
  int e = idx / CONVK, k = idx - e * CONVK;
  Wt[idx] = f2bf(k < 1200 ? W[e * 1200 + k] : 0.0f);
}

// ---------------------------------------------------------------------------
// Main bf16 WMMA GEMM: C(M,N) = A(M,K) * Wt(N,K)^T  (+ epilogue)
// EPI 0: t = C + bias + pos_embed      (conv embed, CONV_A im2col)
// EPI 1: obf = bf16(C + bias)          (qkv)
// EPI 2: t += C + bias                 (proj / fc2 residual add)
// EPI 3: obf = bf16(gelu(C + bias))    (fc1)
// Block 256 thr = 8 waves (4M x 2N); BM=BN=128, BK=32; wave tile 32x64.
// Dense tiles staged via global_load_async_to_lds_b128 (ASYNCcnt overlap).
// ---------------------------------------------------------------------------
template <int EPI, bool CONV_A>
__global__ __launch_bounds__(256) void gemm_bf16(
    const u16* __restrict__ A, const float* __restrict__ X,
    const u16* __restrict__ Wt, const float* __restrict__ bias,
    const float* __restrict__ pos, float* __restrict__ tres,
    u16* __restrict__ obf, int Mtot, int Ntot, int Kd) {
  __shared__ __align__(16) u16 As[2][128 * 40];
  __shared__ __align__(16) u16 Bs[2][128 * 40];

  const int tid = threadIdx.x;
  const int lane = tid & 31, wave = tid >> 5;
  const int wm = wave & 3, wn = wave >> 2;
  const int bm = blockIdx.y * 128, bn = blockIdx.x * 128;

  const int r = tid >> 1;           // 0..127 : row of tile handled by thread
  const int kc = (tid & 1) << 4;    // 0 or 16 : k-chunk

  v8f acc[2][4];
#pragma unroll
  for (int i = 0; i < 2; i++)
#pragma unroll
    for (int j = 0; j < 4; j++)
#pragma unroll
      for (int e = 0; e < 8; e++) acc[i][j][e] = 0.0f;

  auto stageAB = [&](int buf, int ks) {
    const int k0 = ks * 32;
    // --- B tile: Wt rows contiguous in K; async DMA straight into LDS ---
    {
      const u16* src = Wt + (size_t)(bn + r) * Kd + k0 + kc;
      async_ld_b128(src, &Bs[buf][r * 40 + kc]);
      async_ld_b128(src + 8, &Bs[buf][r * 40 + kc + 8]);
    }
    // --- A tile ---
    if (!CONV_A) {
      int gr = bm + r;
      if (gr >= Mtot) gr = Mtot - 1;
      const u16* src = A + (size_t)gr * Kd + k0 + kc;
      async_ld_b128(src, &As[buf][r * 40 + kc]);
      async_ld_b128(src + 8, &As[buf][r * 40 + kc + 8]);
    } else {
      // im2col on the fly: row=(b,p,q), k = c*100 + k1*10 + k2
      int gr = bm + r;
      int bI = gr / LDIM, l = gr - bI * LDIM;
      int p = l / SDIM, q = l - p * SDIM;
      unsigned pk[8];
#pragma unroll
      for (int kk2 = 0; kk2 < 8; kk2++) {
        unsigned w = 0;
#pragma unroll
        for (int half = 0; half < 2; half++) {
          int k = k0 + kc + kk2 * 2 + half;
          float v = 0.0f;
          if (gr < Mtot && k < 1200) {
            int c = k / 100, rr = k - c * 100;
            int k1 = rr / 10, k2 = rr - k1 * 10;
            int ii = p - 5 + k1, jj = q - 5 + k2;
            if ((unsigned)ii < 40u && (unsigned)jj < 40u)
              v = X[(((size_t)bI * 12 + c) * 40 + ii) * 40 + jj];
          }
          w |= (unsigned)f2bf(v) << (16 * half);
        }
        pk[kk2] = w;
      }
      u32x4 w0 = {pk[0], pk[1], pk[2], pk[3]};
      u32x4 w1 = {pk[4], pk[5], pk[6], pk[7]};
      *(u32x4*)&As[buf][r * 40 + kc] = w0;
      *(u32x4*)&As[buf][r * 40 + kc + 8] = w1;
    }
  };

  const int nK = Kd >> 5;
  stageAB(0, 0);
  wait_async0();
  __syncthreads();

  int cur = 0;
  for (int ks = 0; ks < nK; ks++) {
    if (ks + 1 < nK) stageAB(cur ^ 1, ks + 1);   // DMA overlaps WMMAs below

    const int lm = lane & 15, lh = lane >> 4;
    FragBF af[2], bf[4];
#pragma unroll
    for (int mt = 0; mt < 2; mt++) {
      const u16* p0 = &As[cur][(wm * 32 + mt * 16 + lm) * 40 + lh * 8];
      af[mt].q[0] = *(const u32x4*)p0;        // K {0-7 | 8-15}
      af[mt].q[1] = *(const u32x4*)(p0 + 16); // K {16-23 | 24-31}
    }
#pragma unroll
    for (int nt = 0; nt < 4; nt++) {
      const u16* p0 = &Bs[cur][(wn * 64 + nt * 16 + lm) * 40 + lh * 8];
      bf[nt].q[0] = *(const u32x4*)p0;        // K {0-7 | 8-15}
      bf[nt].q[1] = *(const u32x4*)(p0 + 16); // K {16-23 | 24-31}
    }
#pragma unroll
    for (int mt = 0; mt < 2; mt++)
#pragma unroll
      for (int nt = 0; nt < 4; nt++)
        acc[mt][nt] = __builtin_amdgcn_wmma_f32_16x16x32_bf16(
            false, af[mt].f, false, bf[nt].f, (short)0, acc[mt][nt], false, false);

    wait_async0();
    __syncthreads();
    cur ^= 1;
  }

  // ------------------------- epilogue -------------------------
  const int lm = lane & 15, lh = lane >> 4;
#pragma unroll
  for (int mt = 0; mt < 2; mt++)
#pragma unroll
    for (int nt = 0; nt < 4; nt++)
#pragma unroll
      for (int v = 0; v < 8; v++) {
        int m = bm + wm * 32 + mt * 16 + v + lh * 8;
        int n = bn + wn * 64 + nt * 16 + lm;
        if (m < Mtot) {
          float c = acc[mt][nt][v] + bias[n];
          if (EPI == 0) {
            tres[(size_t)m * Ntot + n] = c + pos[(size_t)(m % LDIM) * Ntot + n];
          } else if (EPI == 1) {
            obf[(size_t)m * Ntot + n] = f2bf(c);
          } else if (EPI == 2) {
            tres[(size_t)m * Ntot + n] += c;
          } else {
            float g = 0.5f * c * (1.0f + erff(c * 0.70710678f));
            obf[(size_t)m * Ntot + n] = f2bf(g);
          }
        }
      }
}

// ---------------------------------------------------------------------------
// Row-axial attention: one wave per (b, s, head). 41 padded to 48.
// qkv: (M, 2304) bf16, row l=(b*1681+s*41+n); q|k|v at col {0,768,1536}+h*64
// ---------------------------------------------------------------------------
__global__ __launch_bounds__(64) void attn_kernel(const u16* __restrict__ qkv,
                                                  u16* __restrict__ o) {
  __shared__ __align__(16) u16 sQ[2][48 * 64];  // Q, later reused for P
  __shared__ __align__(16) u16 sK[2][48 * 64];
  __shared__ __align__(16) u16 sV[2][64 * 64];  // V transposed: [col][key]

  const int lane = threadIdx.x & 31, wave = threadIdx.x >> 5;
  const int h = blockIdx.x * 2 + wave;
  const int s = blockIdx.y, b = blockIdx.z;
  u16* Q = sQ[wave];
  u16* Kt = sK[wave];
  u16* Vt = sV[wave];

  const size_t rowBase = ((size_t)b * LDIM + (size_t)s * SDIM) * QKVDIM;

  for (int n = 0; n < 48; n++) {
    unsigned qv = 0, kv = 0;
    if (n < SDIM) {
      const u16* rp = qkv + rowBase + (size_t)n * QKVDIM + h * HDIM + 2 * lane;
      qv = *(const unsigned*)rp;
      kv = *(const unsigned*)(rp + EDIM);
    }
    *(unsigned*)&Q[n * 64 + 2 * lane] = qv;
    *(unsigned*)&Kt[n * 64 + 2 * lane] = kv;
  }
  for (int n = 0; n < 64; n++) {
    unsigned vv = 0;
    if (n < SDIM)
      vv = *(const unsigned*)(qkv + rowBase + (size_t)n * QKVDIM + 2 * EDIM +
                              h * HDIM + 2 * lane);
    Vt[(2 * lane) * 64 + n] = (u16)(vv & 0xFFFFu);
    Vt[(2 * lane + 1) * 64 + n] = (u16)(vv >> 16);
  }
  __syncthreads();

  const int lm = lane & 15, lh = lane >> 4;

  // ---- scores = Q @ K^T  (48x48, Kdim=64) ----
  v8f sc[3][3];
#pragma unroll
  for (int i = 0; i < 3; i++)
#pragma unroll
    for (int j = 0; j < 3; j++)
#pragma unroll
      for (int e = 0; e < 8; e++) sc[i][j][e] = 0.0f;

#pragma unroll
  for (int ks = 0; ks < 2; ks++) {
    FragBF qa[3], kb[3];
#pragma unroll
    for (int mt = 0; mt < 3; mt++) {
      const u16* p = &Q[(mt * 16 + lm) * 64 + ks * 32 + lh * 8];
      qa[mt].q[0] = *(const u32x4*)p;
      qa[mt].q[1] = *(const u32x4*)(p + 16);
    }
#pragma unroll
    for (int nt = 0; nt < 3; nt++) {
      const u16* p = &Kt[(nt * 16 + lm) * 64 + ks * 32 + lh * 8];
      kb[nt].q[0] = *(const u32x4*)p;
      kb[nt].q[1] = *(const u32x4*)(p + 16);
    }
#pragma unroll
    for (int mt = 0; mt < 3; mt++)
#pragma unroll
      for (int nt = 0; nt < 3; nt++)
        sc[mt][nt] = __builtin_amdgcn_wmma_f32_16x16x32_bf16(
            false, qa[mt].f, false, kb[nt].f, (short)0, sc[mt][nt], false, false);
  }

  // ---- softmax over key dim (cols); rows live in 16-lane halves ----
  const float scale = 0.125f;  // HD^-0.5
#pragma unroll
  for (int mt = 0; mt < 3; mt++) {
#pragma unroll
    for (int v = 0; v < 8; v++) {
      float x0 = sc[mt][0][v] * scale;
      float x1 = sc[mt][1][v] * scale;
      float x2 = sc[mt][2][v] * scale;
      if (16 + lm > 40) x1 = -1e30f;
      if (32 + lm > 40) x2 = -1e30f;
      float mx = fmaxf(x0, fmaxf(x1, x2));
#pragma unroll
      for (int off = 8; off; off >>= 1) mx = fmaxf(mx, __shfl_xor(mx, off, 16));
      float e0 = __expf(x0 - mx), e1 = __expf(x1 - mx), e2 = __expf(x2 - mx);
      float sm = e0 + e1 + e2;
#pragma unroll
      for (int off = 8; off; off >>= 1) sm += __shfl_xor(sm, off, 16);
      float inv = 1.0f / sm;
      sc[mt][0][v] = e0 * inv;
      sc[mt][1][v] = e1 * inv;
      sc[mt][2][v] = e2 * inv;
    }
  }

  __syncthreads();
  // ---- P -> bf16 in LDS (reuse Q); zero pad cols 48..63 ----
#pragma unroll
  for (int mt = 0; mt < 3; mt++)
#pragma unroll
    for (int nt = 0; nt < 3; nt++)
#pragma unroll
      for (int v = 0; v < 8; v++) {
        int m = mt * 16 + v + lh * 8;
        Q[m * 64 + nt * 16 + lm] = f2bf(sc[mt][nt][v]);
      }
  {
    u32x4 zz = {0, 0, 0, 0};
    for (int rr = lane; rr < 48; rr += 32) {
      *(u32x4*)&Q[rr * 64 + 48] = zz;
      *(u32x4*)&Q[rr * 64 + 56] = zz;
    }
  }
  __syncthreads();

  // ---- O = P @ V  (48x64, Kdim=64 with zero pad) ----
  v8f ov[3][4];
#pragma unroll
  for (int i = 0; i < 3; i++)
#pragma unroll
    for (int j = 0; j < 4; j++)
#pragma unroll
      for (int e = 0; e < 8; e++) ov[i][j][e] = 0.0f;

#pragma unroll
  for (int ks = 0; ks < 2; ks++) {
    FragBF pa[3], vb[4];
#pragma unroll
    for (int mt = 0; mt < 3; mt++) {
      const u16* p = &Q[(mt * 16 + lm) * 64 + ks * 32 + lh * 8];
      pa[mt].q[0] = *(const u32x4*)p;
      pa[mt].q[1] = *(const u32x4*)(p + 16);
    }
#pragma unroll
    for (int nt = 0; nt < 4; nt++) {
      const u16* p = &Vt[(nt * 16 + lm) * 64 + ks * 32 + lh * 8];
      vb[nt].q[0] = *(const u32x4*)p;
      vb[nt].q[1] = *(const u32x4*)(p + 16);
    }
#pragma unroll
    for (int mt = 0; mt < 3; mt++)
#pragma unroll
      for (int nt = 0; nt < 4; nt++)
        ov[mt][nt] = __builtin_amdgcn_wmma_f32_16x16x32_bf16(
            false, pa[mt].f, false, vb[nt].f, (short)0, ov[mt][nt], false, false);
  }

  const size_t obase = ((size_t)b * LDIM + (size_t)s * SDIM) * EDIM + h * HDIM;
#pragma unroll
  for (int mt = 0; mt < 3; mt++)
#pragma unroll
    for (int nt = 0; nt < 4; nt++)
#pragma unroll
      for (int v = 0; v < 8; v++) {
        int m = mt * 16 + v + lh * 8;
        if (m < SDIM)
          o[obase + (size_t)m * EDIM + nt * 16 + lm] = f2bf(ov[mt][nt][v]);
      }
}

// ---------------------------------------------------------------------------
// LayerNorm over E=768; one block per row; bf16 or f32 output
// ---------------------------------------------------------------------------
template <bool BF16OUT>
__global__ __launch_bounds__(256) void ln_kernel(const float* __restrict__ x,
                                                 const float* __restrict__ w,
                                                 const float* __restrict__ bsh,
                                                 u16* __restrict__ ybf,
                                                 float* __restrict__ yf) {
  __shared__ float r1[256], r2[256];
  const int row = blockIdx.x, tid = threadIdx.x;
  const float* xr = x + (size_t)row * EDIM;
  float v0 = xr[tid], v1 = xr[tid + 256], v2 = xr[tid + 512];
  r1[tid] = v0 + v1 + v2;
  r2[tid] = v0 * v0 + v1 * v1 + v2 * v2;
  __syncthreads();
  for (int off = 128; off; off >>= 1) {
    if (tid < off) {
      r1[tid] += r1[tid + off];
      r2[tid] += r2[tid + off];
    }
    __syncthreads();
  }
  float mu = r1[0] * (1.0f / 768.0f);
  float var = r2[0] * (1.0f / 768.0f) - mu * mu;
  float rs = rsqrtf(var + 1e-6f);
#pragma unroll
  for (int i = 0; i < 3; i++) {
    int c = tid + i * 256;
    float xv = (i == 0) ? v0 : (i == 1) ? v1 : v2;
    float val = (xv - mu) * rs * w[c] + bsh[c];
    if (BF16OUT)
      ybf[(size_t)row * EDIM + c] = f2bf(val);
    else
      yf[(size_t)row * EDIM + c] = val;
  }
}

// ---------------------------------------------------------------------------
// Output transposed-conv (as conv with flipped weights, pad=4): 2 GFLOP, VALU
// ---------------------------------------------------------------------------
__global__ __launch_bounds__(256) void tconv_kernel(const float* __restrict__ feat,
                                                    const float* __restrict__ tw,
                                                    const float* __restrict__ tb,
                                                    float* __restrict__ out) {
  __shared__ float red[256];
  const int bid = blockIdx.x;
  const int b = bid / 1600, rem = bid - b * 1600;
  const int i = rem / 40, j = rem - i * 40;
  const int tid = threadIdx.x;
  float acc = 0.0f;
  for (int a = 0; a < 10; a++) {
    int p = i - 4 + a;
    if ((unsigned)p >= (unsigned)SDIM) continue;
    for (int c2 = 0; c2 < 10; c2++) {
      int q = j - 4 + c2;
      if ((unsigned)q >= (unsigned)SDIM) continue;
      const float* fr = feat + ((size_t)b * LDIM + p * SDIM + q) * EDIM;
      const int widx = (9 - a) * 10 + (9 - c2);
      for (int e = tid; e < EDIM; e += 256) acc += fr[e] * tw[e * 100 + widx];
    }
  }
  red[tid] = acc;
  __syncthreads();
  for (int off = 128; off; off >>= 1) {
    if (tid < off) red[tid] += red[tid + off];
    __syncthreads();
  }
  if (tid == 0) out[bid] = red[0] + tb[0];
}

// ---------------------------------------------------------------------------
extern "C" void kernel_launch(void* const* d_in, const int* in_sizes, int n_in,
                              void* d_out, int out_size, void* d_ws, size_t ws_size,
                              hipStream_t stream) {
  const float* x      = (const float*)d_in[0];
  const float* conv_w = (const float*)d_in[1];
  const float* conv_b = (const float*)d_in[2];
  const float* pos    = (const float*)d_in[3];
  const float* ln1w   = (const float*)d_in[4];
  const float* ln1b   = (const float*)d_in[5];
  const float* qkvw   = (const float*)d_in[6];
  const float* qkvb   = (const float*)d_in[7];
  const float* projw  = (const float*)d_in[8];
  const float* projb  = (const float*)d_in[9];
  const float* ln2w   = (const float*)d_in[10];
  const float* ln2b   = (const float*)d_in[11];
  const float* fc1w   = (const float*)d_in[12];
  const float* fc1b   = (const float*)d_in[13];
  const float* fc2w   = (const float*)d_in[14];
  const float* fc2b   = (const float*)d_in[15];
  const float* lnfw   = (const float*)d_in[16];
  const float* lnfb   = (const float*)d_in[17];
  const float* tcw    = (const float*)d_in[18];
  const float* tcb    = (const float*)d_in[19];
  float* out = (float*)d_out;

  char* ws = (char*)d_ws;
  auto al = [](size_t v) { return (v + 255) & ~(size_t)255; };
  size_t off = 0;
  float* t = (float*)(ws + off);      off = al(off + (size_t)MROWS * EDIM * 4);
  u16* act16 = (u16*)(ws + off);      off = al(off + (size_t)MROWS * EDIM * 2);
  u16* big = (u16*)(ws + off);        off = al(off + (size_t)MROWS * MLPDIM * 2);
  u16* convWt = (u16*)(ws + off);     off = al(off + (size_t)EDIM * CONVK * 2);
  u16* qkvWt = (u16*)(ws + off);      off = al(off + (size_t)12 * QKVDIM * EDIM * 2);
  u16* projWt = (u16*)(ws + off);     off = al(off + (size_t)12 * EDIM * EDIM * 2);
  u16* fc1Wt = (u16*)(ws + off);      off = al(off + (size_t)12 * MLPDIM * EDIM * 2);
  u16* fc2Wt = (u16*)(ws + off);      off = al(off + (size_t)12 * EDIM * MLPDIM * 2);
  float* feat = (float*)big;  // reuse: big is dead after last fc2

  const int mBlocks = (MROWS + 127) / 128;  // 211

  // ---- weight conversion (bf16, pre-transposed N x K) ----
  {
    long tot = (long)EDIM * CONVK;
    convwt_convert<<<dim3((unsigned)((tot + 255) / 256)), 256, 0, stream>>>(conv_w, convWt);
  }
  {
    long tot = 12L * QKVDIM * EDIM;
    wt_convert<<<dim3((unsigned)((tot + 255) / 256)), 256, 0, stream>>>(qkvw, qkvWt, EDIM, QKVDIM, tot);
  }
  {
    long tot = 12L * EDIM * EDIM;
    wt_convert<<<dim3((unsigned)((tot + 255) / 256)), 256, 0, stream>>>(projw, projWt, EDIM, EDIM, tot);
  }
  {
    long tot = 12L * MLPDIM * EDIM;
    wt_convert<<<dim3((unsigned)((tot + 255) / 256)), 256, 0, stream>>>(fc1w, fc1Wt, EDIM, MLPDIM, tot);
  }
  {
    long tot = 12L * EDIM * MLPDIM;
    wt_convert<<<dim3((unsigned)((tot + 255) / 256)), 256, 0, stream>>>(fc2w, fc2Wt, MLPDIM, EDIM, tot);
  }

  // ---- patch-embed conv as im2col GEMM: t = x*W + b + pos ----
  gemm_bf16<0, true><<<dim3(EDIM / 128, mBlocks), 256, 0, stream>>>(
      nullptr, x, convWt, conv_b, pos, t, nullptr, MROWS, EDIM, CONVK);

  for (int i = 0; i < 12; i++) {
    ln_kernel<true><<<dim3(MROWS), 256, 0, stream>>>(t, ln1w + i * EDIM, ln1b + i * EDIM, act16, nullptr);
    gemm_bf16<1, false><<<dim3(QKVDIM / 128, mBlocks), 256, 0, stream>>>(
        act16, nullptr, qkvWt + (size_t)i * QKVDIM * EDIM, qkvb + i * QKVDIM,
        nullptr, nullptr, big, MROWS, QKVDIM, EDIM);
    attn_kernel<<<dim3(NHEAD / 2, SDIM, BV), 64, 0, stream>>>(big, act16);
    gemm_bf16<2, false><<<dim3(EDIM / 128, mBlocks), 256, 0, stream>>>(
        act16, nullptr, projWt + (size_t)i * EDIM * EDIM, projb + i * EDIM,
        nullptr, t, nullptr, MROWS, EDIM, EDIM);
    ln_kernel<true><<<dim3(MROWS), 256, 0, stream>>>(t, ln2w + i * EDIM, ln2b + i * EDIM, act16, nullptr);
    gemm_bf16<3, false><<<dim3(MLPDIM / 128, mBlocks), 256, 0, stream>>>(
        act16, nullptr, fc1Wt + (size_t)i * MLPDIM * EDIM, fc1b + i * MLPDIM,
        nullptr, nullptr, big, MROWS, MLPDIM, EDIM);
    gemm_bf16<2, false><<<dim3(EDIM / 128, mBlocks), 256, 0, stream>>>(
        big, nullptr, fc2Wt + (size_t)i * EDIM * MLPDIM, fc2b + i * EDIM,
        nullptr, t, nullptr, MROWS, EDIM, MLPDIM);
  }

  ln_kernel<false><<<dim3(MROWS), 256, 0, stream>>>(t, lnfw, lnfb, nullptr, feat);
  tconv_kernel<<<dim3(BV * 40 * 40), 256, 0, stream>>>(feat, tcw, tcb, out);
}
